// NBA_25778393710811
// MI455X (gfx1250) — matrix operand order repaired and checked by reading, not compiled, not run on hardware
//
#include <hip/hip_runtime.h>
#include <hip/hip_bf16.h>
#include <math.h>

typedef __attribute__((ext_vector_type(16))) _Float16 v16h;
typedef __attribute__((ext_vector_type(8)))  _Float16 h8;
typedef __attribute__((ext_vector_type(8)))  float    v8f;

#define T_STEPS 512
#define HID     64
#define G4      256
#define MT      16

// Branchless, hardware-native activations (v_exp_f32 is exp2 on AMD).
__device__ __forceinline__ float sigm(float v) {
  return __builtin_amdgcn_rcpf(1.0f + __builtin_amdgcn_exp2f(v * -1.44269504088896340736f));
}
__device__ __forceinline__ float tanh_fast(float v) {
  return 2.0f * sigm(2.0f * v) - 1.0f;
}

// D = A(16xK f16) x B(Kx16 f16) + C(f32), one 16x16x32 WMMA
__device__ __forceinline__ v8f wmma16(v16h a, v16h b, v8f c) {
  return __builtin_amdgcn_wmma_f32_16x16x32_f16(
      /*neg_a=*/false, a, /*neg_b=*/false, b,
      /*c_mod=*/(short)0, c, /*reuse_a=*/false, /*reuse_b=*/false);
}

// A fragment (16x32, f16) from LDS. Lane L: row m = L&15.
// per-lane elements j=0..15 -> K = 16*hi + j
__device__ __forceinline__ v16h load_afrag(const _Float16* p, int m, int stride,
                                           int koff, int hi) {
  const h8* q0 = (const h8*)(p + m * stride + koff + hi * 16);
  const h8* q1 = (const h8*)(p + m * stride + koff + hi * 16 + 8);
  h8 r0 = *q0;
  h8 r1 = *q1;
  return __builtin_shufflevector(r0, r1, 0, 1, 2, 3, 4, 5, 6, 7,
                                 8, 9, 10, 11, 12, 13, 14, 15);
}

// B fragment (32x16, f16) built from global f32 weights W[256][srcK], row-major.
// B[k][n] = W[ng][k]; lane L: n = L&15, K = 16*hi + j  (j contiguous in row).
__device__ __forceinline__ v16h load_bfrag(const float* __restrict__ src, int srcK,
                                           int koff, int ng, int kk) {
  v16h b;
#pragma unroll
  for (int j = 0; j < 16; ++j) {
    int k = koff + kk + j;
    float v = (k < srcK) ? src[ng * srcK + k] : 0.0f;
    b[j] = (_Float16)v;
  }
  return b;
}

// Broadcast per-column bias into all 8 accumulator rows (C operand of WMMA).
__device__ __forceinline__ v8f bias_acc(float b) {
  v8f a;
#pragma unroll
  for (int r = 0; r < 8; ++r) a[r] = b;
  return a;
}

__global__ __launch_bounds__(512) void lstm3_fused(
    const float* __restrict__ x,
    const float* __restrict__ wih0, const float* __restrict__ whh0,
    const float* __restrict__ bih0, const float* __restrict__ bhh0,
    const float* __restrict__ wih1, const float* __restrict__ whh1,
    const float* __restrict__ bih1, const float* __restrict__ bhh1,
    const float* __restrict__ wih2, const float* __restrict__ whh2,
    const float* __restrict__ bih2, const float* __restrict__ bhh2,
    const float* __restrict__ wlin, const float* __restrict__ blin,
    float* __restrict__ out) {
  __shared__ _Float16 s_x[MT * 32] __attribute__((aligned(16)));      // x_t, K-padded
  __shared__ _Float16 s_h[3][MT * HID] __attribute__((aligned(16)));  // per-layer h
  __shared__ float s_g[MT * G4];                                      // gate buffer (bias folded in)
  __shared__ float s_h2f[MT * HID];                                   // layer-2 h, f32

  const int tid  = threadIdx.x;
  const int lane = tid & 31;
  const int wv   = tid >> 5;       // wave 0..15 -> gate column tile
  const int nl   = lane & 15;
  const int hi   = lane >> 4;      // 0/1 half of the wave
  const int kk   = hi * 16;        // B-fragment K base
  const int b0   = blockIdx.x * MT;
  const int ng   = wv * 16 + nl;   // gate column 0..255 owned by this lane

  // ---- one-time init: zero h state ----
  {
    _Float16* hflat = &s_h[0][0];
    for (int i = tid; i < 3 * MT * HID; i += 512) hflat[i] = (_Float16)0.0f;
  }

  // ---- per-lane combined bias for this lane's gate column (rides WMMA C operand) ----
  float bs0 = bih0[ng] + bhh0[ng];
  float bs1 = bih1[ng] + bhh1[ng];
  float bs2 = bih2[ng] + bhh2[ng];

  // ---- preload all B fragments (time-invariant weights) into VGPRs ----
  v16h bfr[3][4];
  bfr[0][0] = load_bfrag(wih0, 24, 0,  ng, kk);   // x chunk (padded 24->32)
  bfr[0][1] = load_bfrag(whh0, 64, 0,  ng, kk);   // h chunk 0
  bfr[0][2] = load_bfrag(whh0, 64, 32, ng, kk);   // h chunk 1
  bfr[0][3] = bfr[0][2];                          // unused
  bfr[1][0] = load_bfrag(wih1, 64, 0,  ng, kk);
  bfr[1][1] = load_bfrag(wih1, 64, 32, ng, kk);
  bfr[1][2] = load_bfrag(whh1, 64, 0,  ng, kk);
  bfr[1][3] = load_bfrag(whh1, 64, 32, ng, kk);
  bfr[2][0] = load_bfrag(wih2, 64, 0,  ng, kk);
  bfr[2][1] = load_bfrag(wih2, 64, 32, ng, kk);
  bfr[2][2] = load_bfrag(whh2, 64, 0,  ng, kk);
  bfr[2][3] = load_bfrag(whh2, 64, 32, ng, kk);

  float creg[3][2];
#pragma unroll
  for (int l = 0; l < 3; ++l) { creg[l][0] = 0.0f; creg[l][1] = 0.0f; }

  // Per-thread x staging pointer: thread handles element (m = tid>>5, k = tid&31)
  const int  xm     = tid >> 5;
  const int  xk     = tid & 31;
  const bool xvalid = (xk < 24);
  const float* xptr = x + (size_t)(b0 + xm) * (size_t)T_STEPS * 24 + (xvalid ? xk : 0);

  __syncthreads();

  // ---- recurrent scan ----
  for (int t = 0; t < T_STEPS; ++t) {
    // stage x_t -> f16, zero-padded to K=32 (512 elements, 1 per thread)
    {
      float v = xvalid ? *xptr : 0.0f;
      s_x[tid] = (_Float16)v;
      xptr += 24;
    }
    __syncthreads();

#pragma unroll
    for (int l = 0; l < 3; ++l) {
      // gates: G[16 x 256] = [x_t | h_{l,t-1}] x W^T + bias  (this wave: 16 columns)
      // Hoist all A-fragment LDS loads so they pipeline; C initialized with bias.
      v8f acc;
      if (l == 0) {
        v16h a0 = load_afrag(s_x,    nl, 32, 0,  hi);
        v16h a1 = load_afrag(s_h[0], nl, 64, 0,  hi);
        v16h a2 = load_afrag(s_h[0], nl, 64, 32, hi);
        acc = bias_acc(bs0);
        acc = wmma16(a0, bfr[0][0], acc);
        acc = wmma16(a1, bfr[0][1], acc);
        acc = wmma16(a2, bfr[0][2], acc);
      } else if (l == 1) {
        v16h a0 = load_afrag(s_h[0], nl, 64, 0,  hi);
        v16h a1 = load_afrag(s_h[0], nl, 64, 32, hi);
        v16h a2 = load_afrag(s_h[1], nl, 64, 0,  hi);
        v16h a3 = load_afrag(s_h[1], nl, 64, 32, hi);
        acc = bias_acc(bs1);
        acc = wmma16(a0, bfr[1][0], acc);
        acc = wmma16(a1, bfr[1][1], acc);
        acc = wmma16(a2, bfr[1][2], acc);
        acc = wmma16(a3, bfr[1][3], acc);
      } else {
        v16h a0 = load_afrag(s_h[1], nl, 64, 0,  hi);
        v16h a1 = load_afrag(s_h[1], nl, 64, 32, hi);
        v16h a2 = load_afrag(s_h[2], nl, 64, 0,  hi);
        v16h a3 = load_afrag(s_h[2], nl, 64, 32, hi);
        acc = bias_acc(bs2);
        acc = wmma16(a0, bfr[2][0], acc);
        acc = wmma16(a1, bfr[2][1], acc);
        acc = wmma16(a2, bfr[2][2], acc);
        acc = wmma16(a3, bfr[2][3], acc);
      }
      // scatter D: VGPR r -> row m = 8*hi + r, col = ng
#pragma unroll
      for (int r = 0; r < 8; ++r) {
        int m = hi * 8 + r;
        s_g[m * G4 + ng] = acc[r];
      }
      __syncthreads();

      // fused LSTM cell: 1024 (m,n) elements, 2 per thread; c lives in registers
#pragma unroll
      for (int k2 = 0; k2 < 2; ++k2) {
        int e = tid * 2 + k2;
        int m = e >> 6, n = e & 63;
        float gi = s_g[m * G4 + n];
        float gf = s_g[m * G4 + 64 + n];
        float gg = s_g[m * G4 + 128 + n];
        float go = s_g[m * G4 + 192 + n];
        float c  = sigm(gf) * creg[l][k2] + sigm(gi) * tanh_fast(gg);
        creg[l][k2] = c;
        float h  = sigm(go) * tanh_fast(c);
        s_h[l][m * HID + n] = (_Float16)h;
        if (l == 2) s_h2f[m * HID + n] = h;
      }
      __syncthreads();
    }
  }

  // ---- final linear head on h2[:, T-1]: out[16 x 4] ----
  if (tid < MT * 4) {
    int m = tid >> 2, of = tid & 3;
    float s = blin[of];
#pragma unroll 8
    for (int n = 0; n < HID; ++n) s += s_h2f[m * HID + n] * wlin[of * HID + n];
    out[(size_t)(b0 + m) * 4 + of] = s;
  }
}

extern "C" void kernel_launch(void* const* d_in, const int* in_sizes, int n_in,
                              void* d_out, int out_size, void* d_ws, size_t ws_size,
                              hipStream_t stream) {
  (void)in_sizes; (void)n_in; (void)d_ws; (void)ws_size; (void)out_size;
  const float* x    = (const float*)d_in[0];
  const float* wih0 = (const float*)d_in[1];
  const float* whh0 = (const float*)d_in[2];
  const float* bih0 = (const float*)d_in[3];
  const float* bhh0 = (const float*)d_in[4];
  const float* wih1 = (const float*)d_in[5];
  const float* whh1 = (const float*)d_in[6];
  const float* bih1 = (const float*)d_in[7];
  const float* bhh1 = (const float*)d_in[8];
  const float* wih2 = (const float*)d_in[9];
  const float* whh2 = (const float*)d_in[10];
  const float* bih2 = (const float*)d_in[11];
  const float* bhh2 = (const float*)d_in[12];
  const float* wlin = (const float*)d_in[13];
  const float* blin = (const float*)d_in[14];

  lstm3_fused<<<dim3(1024 / MT), dim3(512), 0, stream>>>(
      x, wih0, whh0, bih0, bhh0, wih1, whh1, bih1, bhh1,
      wih2, whh2, bih2, bhh2, wlin, blin, (float*)d_out);
}
